// CaptionGenerator_2611340116190
// MI455X (gfx1250) — compile-verified, matrix operand order checked by reading, not run on hardware
//
#include <hip/hip_runtime.h>

typedef __attribute__((ext_vector_type(16))) _Float16 v16h;
typedef __attribute__((ext_vector_type(8)))  float    v8f;

#define B_  64
#define T_  21
#define TD  20
#define V_  12000
#define E_  256
#define H_  512
#define D_  1024
#define R_  36
#define KX  1792   // E + D + H concatenated LSTM input

// ---------------------------------------------------------------------------
// CDNA5 async copy helpers (inline asm per CDNA5 ISA --
// portable across toolchains, bypasses the arity-unstable clang builtins).
// ---------------------------------------------------------------------------
__device__ __forceinline__ void async_load_b128(unsigned lds_off, const void* gaddr) {
    asm volatile("global_load_async_to_lds_b128 %0, %1, off"
                 :: "v"(lds_off), "v"((unsigned long long)(size_t)gaddr)
                 : "memory");
}
__device__ __forceinline__ void wait_async0() {
    asm volatile("s_wait_asynccnt 0" ::: "memory");
}

__device__ __forceinline__ v8f wmma_step(const _Float16* a, const _Float16* b, v8f acc) {
    v16h av = *(const v16h*)a;
    v16h bv = *(const v16h*)b;
    return __builtin_amdgcn_wmma_f32_16x16x32_f16(false, av, false, bv,
                                                  (short)0, acc, false, false);
}

// ---------------------------------------------------------------------------
// Sort batch by caption length (descending, stable), emit int outputs.
// ---------------------------------------------------------------------------
__global__ void k_sort(const int* __restrict__ lens, const int* __restrict__ caps,
                       int* sort_ind, int* dec, int* caps_s,
                       float* out_caps, float* out_dec, float* out_si) {
    int i = threadIdx.x;
    if (i >= B_) return;
    int li = lens[i];
    int rank = 0;
    for (int j = 0; j < B_; ++j) {
        int lj = lens[j];
        if (lj > li || (lj == li && j < i)) rank++;
    }
    sort_ind[rank] = i;
    __syncthreads();
    int src = sort_ind[i];
    int d = lens[src] - 1;
    dec[i] = d;
    out_dec[i] = (float)d;
    out_si[i] = (float)src;
    for (int k = 0; k < T_; ++k) caps_s[i*T_ + k] = caps[src*T_ + k];
    for (int k = 0; k < TD; ++k) out_caps[i*TD + k] = (float)caps[src*T_ + k + 1];
}

// ---------------------------------------------------------------------------
// Convert weights to f16, build concatenated LSTM weight, zero colsum.
// ---------------------------------------------------------------------------
__global__ void k_convert(const float* embW, const float* whW, const float* wrW,
                          const float* rhW, const float* Wih, const float* Whh,
                          const float* bih, const float* bhh,
                          _Float16* embh, _Float16* whh16, _Float16* wrh,
                          _Float16* rhwh, _Float16* Wcat, float* bcat, float* colsum) {
    long i = (long)blockIdx.x * 256 + threadIdx.x;
    if (i < (long)V_*E_) embh[i] = (_Float16)embW[i];
    if (i < H_*E_)       whh16[i] = (_Float16)whW[i];
    if (i < D_*E_)       wrh[i]   = (_Float16)wrW[i];
    if (i < H_*D_)       rhwh[i]  = (_Float16)rhW[i];
    if (i < (long)4*H_*KX) {
        int j = (int)(i / KX), kk = (int)(i % KX);
        float v = (kk < E_ + D_) ? Wih[(long)j*(E_+D_) + kk]
                                 : Whh[(long)j*H_ + (kk - (E_+D_))];
        Wcat[i] = (_Float16)v;
    }
    if (i < 4*H_) bcat[i] = bih[i] + bhh[i];
    if (i < D_)   colsum[i] = 0.f;
}

// ---------------------------------------------------------------------------
// Gather sorted objs/h0, initialize h,c,rf and LSTM input row xh.
// ---------------------------------------------------------------------------
__global__ void k_gather(const float* h0, const float* objs, const int* sort_ind,
                         float* objs_s, _Float16* objs_h, float* h, float* c,
                         float* rf, _Float16* xh, _Float16* hh16) {
    long i = (long)blockIdx.x * 256 + threadIdx.x;
    if (i < (long)B_*R_*D_) {
        int b = (int)(i / (R_*D_));
        long rd = i % (R_*D_);
        float v = objs[(long)sort_ind[b]*R_*D_ + rd];
        objs_s[i] = v;
        objs_h[i] = (_Float16)v;
    }
    if (i < B_*H_) {
        int b = (int)(i / H_), u = (int)(i % H_);
        float v = h0[(long)sort_ind[b]*H_ + u];
        h[i] = v; c[i] = v;
        hh16[i] = (_Float16)v;
        xh[(long)b*KX + (E_+D_) + u] = (_Float16)v;
    }
    if (i < B_*D_) {
        int b = (int)(i / D_), d = (int)(i % D_);
        rf[i] = 0.f;
        xh[(long)b*KX + E_ + d] = (_Float16)0.f;
    }
}

__global__ void k_objsum(const float* objs_s, _Float16* obj_sum) {
    int i = blockIdx.x * 256 + threadIdx.x;   // B*D
    int b = i / D_, d = i % D_;
    float s = 0.f;
    for (int r = 0; r < R_; ++r) s += objs_s[((long)b*R_ + r)*D_ + d];
    obj_sum[i] = (_Float16)s;
}

__global__ void k_pw(const float* embW, const float* wW, const float* wb, float* Pw) {
    int v = blockIdx.x * 256 + threadIdx.x;
    if (v >= V_) return;
    float s = wb[0];
    for (int e = 0; e < E_; ++e) s += embW[(long)v*E_ + e] * wW[e];
    Pw[v] = s;
}

// ---------------------------------------------------------------------------
// Generic WMMA GEMM: out(M x N) = A(M x K) * B(N x K)^T.
// Each wave owns a 16 x (16*STRIP) strip: one A load feeds STRIP WMMAs.
// ---------------------------------------------------------------------------
template<int STRIP, bool F16OUT>
__global__ void k_gemm(const _Float16* __restrict__ A, int lda,
                       const _Float16* __restrict__ Bm, int ldb, int K,
                       void* __restrict__ outv, int ldo,
                       const float* __restrict__ bias, float* colsum,
                       int nstrips, int total) {
    int wid = blockIdx.x * (blockDim.x >> 5) + (threadIdx.x >> 5);
    if (wid >= total) return;
    int mt = wid / nstrips, st = wid % nstrips;
    int lane = threadIdx.x & 31;
    int half = lane >> 4, idx = lane & 15;
    const _Float16* ar = A + (long)(mt*16 + idx)*lda + 16*half;
    const _Float16* br[STRIP];
#pragma unroll
    for (int s = 0; s < STRIP; ++s)
        br[s] = Bm + (long)((st*STRIP + s)*16 + idx)*ldb + 16*half;
    v8f acc[STRIP];
#pragma unroll
    for (int s = 0; s < STRIP; ++s) acc[s] = (v8f){};
    for (int k = 0; k < K; k += 32) {
        __builtin_prefetch(br[0] + k + 512, 0, 0);   // global_prefetch_b8
        // issue all loads first so WMMAs retire under partial load counts
        v16h a = *(const v16h*)(ar + k);
        v16h bv[STRIP];
#pragma unroll
        for (int s = 0; s < STRIP; ++s) bv[s] = *(const v16h*)(br[s] + k);
#pragma unroll
        for (int s = 0; s < STRIP; ++s)
            acc[s] = __builtin_amdgcn_wmma_f32_16x16x32_f16(
                false, a, false, bv[s], (short)0, acc[s], false, false);
    }
#pragma unroll
    for (int s = 0; s < STRIP; ++s) {
        int n = (st*STRIP + s)*16 + idx;
        float bv = bias ? bias[n] : 0.f;
        float cs = 0.f;
#pragma unroll
        for (int v = 0; v < 8; ++v) {
            int m = mt*16 + 8*half + v;
            float val = acc[s][v] + bv;
            if constexpr (F16OUT) ((_Float16*)outv)[(long)m*ldo + n] = (_Float16)val;
            else                  ((float*)outv)[(long)m*ldo + n] = val;
            cs += val;
        }
        if (colsum) atomicAdd(&colsum[n], cs);
    }
}

// r_lin and sumv_wr (needs colsum of P_wr, produced by k_gemm atomics)
__global__ void k_rlin_sumv(const float* objs_s, const float* rW, const float* rb,
                            const float* colsum, float* r_lin, float* sumv) {
    int i = blockIdx.x * 256 + threadIdx.x;    // B*R
    if (i >= B_*R_) return;
    float rl = rb[0], sv = 0.f;
    const float* row = objs_s + (long)i * D_;
    for (int d = 0; d < D_; ++d) {
        float o = row[d];
        rl += o * rW[d];
        sv += o * colsum[d];
    }
    r_lin[i] = rl;
    sumv[i]  = sv;
}

// ---------------------------------------------------------------------------
// Per-step: fill embedding + h slots of xh, zero step accumulators.
// ---------------------------------------------------------------------------
__global__ void k_prep(const int* caps_s, const _Float16* embh, const _Float16* hh16,
                       _Float16* xh, float* wtsum, float* rvsum, int t) {
    int i = blockIdx.x * 256 + threadIdx.x;    // covers B*H = 32768
    if (i < B_*E_) {
        int b = i >> 8, e = i & (E_-1);
        int tok = caps_s[b*T_ + t];
        xh[(long)b*KX + e] = embh[(long)tok*E_ + e];
    }
    if (i < B_*H_) {
        int b = i / H_, u = i % H_;
        xh[(long)b*KX + (E_+D_) + u] = hh16[i];
    }
    if (i < B_) { wtsum[i] = 0.f; rvsum[i] = 0.f; }
}

// ---------------------------------------------------------------------------
// Fused LSTM step: 4 gate tiles (shared A load) + pointwise update + mask.
// All 10 b128 loads per K-step issued up front (one clause), K-loop unrolled
// x2 so WMMAs overlap the next iteration's loads. 128 tiles (4 bt x 32 jt).
// ---------------------------------------------------------------------------
__global__ void k_lstm(const _Float16* __restrict__ xh,
                       const _Float16* __restrict__ Wcat,
                       const float* __restrict__ bcat,
                       float* h, float* c, _Float16* hh16,
                       const int* dec, int t) {
    int wid = blockIdx.x * (blockDim.x >> 5) + (threadIdx.x >> 5);  // 0..127
    int bt = wid >> 5;       // 0..3
    int jt = wid & 31;       // 0..31
    int lane = threadIdx.x & 31;
    int half = lane >> 4, idx = lane & 15;
    const _Float16* ar = xh   + (long)(bt*16 + idx)*KX + 16*half;
    const _Float16* bi = Wcat + (long)(jt*16 + idx)*KX + 16*half;
    const _Float16* bf = bi + (long)H_*KX;
    const _Float16* bg = bi + (long)2*H_*KX;
    const _Float16* bo = bi + (long)3*H_*KX;
    v8f ai = {}, af = {}, ag = {}, ao = {};
    for (int k = 0; k < KX; k += 64) {
#pragma unroll
        for (int kk = 0; kk < 64; kk += 32) {
            v16h a  = *(const v16h*)(ar + k + kk);
            v16h vi = *(const v16h*)(bi + k + kk);
            v16h vf = *(const v16h*)(bf + k + kk);
            v16h vg = *(const v16h*)(bg + k + kk);
            v16h vo = *(const v16h*)(bo + k + kk);
            ai = __builtin_amdgcn_wmma_f32_16x16x32_f16(false, a, false, vi, (short)0, ai, false, false);
            af = __builtin_amdgcn_wmma_f32_16x16x32_f16(false, a, false, vf, (short)0, af, false, false);
            ag = __builtin_amdgcn_wmma_f32_16x16x32_f16(false, a, false, vg, (short)0, ag, false, false);
            ao = __builtin_amdgcn_wmma_f32_16x16x32_f16(false, a, false, vo, (short)0, ao, false, false);
        }
    }
    int u = jt*16 + idx;
    float b_i = bcat[u], b_f = bcat[H_+u], b_g = bcat[2*H_+u], b_o = bcat[3*H_+u];
#pragma unroll
    for (int v = 0; v < 8; ++v) {
        int bb = bt*16 + 8*half + v;
        float gi = ai[v] + b_i, gf = af[v] + b_f, gg = ag[v] + b_g, go = ao[v] + b_o;
        float iv = 1.f / (1.f + expf(-gi));
        float fv = 1.f / (1.f + expf(-gf));
        float ov = 1.f / (1.f + expf(-go));
        float gv = tanhf(gg);
        long ci = (long)bb*H_ + u;
        float cold = c[ci], hold = h[ci];
        float cn = fv*cold + iv*gv;
        float hn = ov*tanhf(cn);
        bool m = dec[bb] > t;
        float cv = m ? cn : cold;
        float hv = m ? hn : hold;
        c[ci] = cv; h[ci] = hv;
        hh16[ci] = (_Float16)hv;
    }
}

// rvec[b,r] = <rh_all[b,r,:], h[b,:]> + r_lin ; rvsum via atomics
__global__ void k_rvec(const float* rh_all, const float* h, const float* r_lin,
                       float* rvec, float* rvsum) {
    int i = blockIdx.x * 256 + threadIdx.x;   // B*R
    if (i >= B_*R_) return;
    int b = i / R_;
    const float* rr = rh_all + (long)i * H_;
    const float* hr = h + (long)b * H_;
    float s = r_lin[i];
    for (int u = 0; u < H_; ++u) s += rr[u] * hr[u];
    rvec[i] = s;
    atomicAdd(&rvsum[b], s);
}

// ---------------------------------------------------------------------------
// wt = h @ P_wh^T (+P_w). The 16x512 A tile (16 KB) is staged once per block
// into LDS via global_load_async_to_lds_b128 (+ s_wait_asynccnt), then all 4
// waves consume it from LDS while streaming B (P_wh) from global/L2.
// Fused: preds store (+sumr_wr +rvsum, masked) and per-batch wt row-sum
// reduction (shfl + atomics) for the attention logits.
// grid = dim3(94, 4), block = 128 (4 waves x 2-tile strips -> 375 strips/mt).
// ---------------------------------------------------------------------------
__global__ void __launch_bounds__(128) k_wt(
        const _Float16* __restrict__ hh16,
        const _Float16* __restrict__ Pwh,
        const float* __restrict__ Pw,
        const float* __restrict__ sumr,
        const float* __restrict__ rvsum,
        float* wtsum, const int* dec, int t,
        float* __restrict__ outP) {
    __shared__ __align__(32) _Float16 As[16 * H_];   // 16 KB
    int mt = blockIdx.y;                             // 0..3
    {   // cooperative async stage: 16*512 halves = 1024 x b128
        const _Float16* Ab = hh16 + (long)mt*16*H_;
        unsigned lb = (unsigned)(size_t)(&As[0]);
        for (int j = threadIdx.x; j < (16*H_)/8; j += 128)
            async_load_b128(lb + j*16u, Ab + j*8);
        wait_async0();
    }
    __syncthreads();
    int strip = blockIdx.x*4 + (threadIdx.x >> 5);   // 0..375 (375 real)
    if (strip >= (V_/16)/2) return;                  // wave-uniform
    int lane = threadIdx.x & 31;
    int half = lane >> 4, idx = lane & 15;
    const _Float16* ar = As + idx*H_ + 16*half;      // LDS (ds_load)
    int nt0 = strip*2;
    const _Float16* b0 = Pwh + (long)(nt0*16 + idx)*H_ + 16*half;
    const _Float16* b1 = b0 + (long)16*H_;
    v8f a0 = {}, a1 = {};
    for (int k = 0; k < H_; k += 32) {
        __builtin_prefetch(b0 + k + 512, 0, 0);      // global_prefetch_b8
        v16h a  = *(const v16h*)(ar + k);
        v16h v0 = *(const v16h*)(b0 + k);
        v16h v1 = *(const v16h*)(b1 + k);
        a0 = __builtin_amdgcn_wmma_f32_16x16x32_f16(false, a, false, v0, (short)0, a0, false, false);
        a1 = __builtin_amdgcn_wmma_f32_16x16x32_f16(false, a, false, v1, (short)0, a1, false, false);
    }
    float wv[8];
#pragma unroll
    for (int v = 0; v < 8; ++v) wv[v] = 0.f;
#pragma unroll
    for (int s = 0; s < 2; ++s) {
        int n = (nt0 + s)*16 + idx;
        float pw = Pw[n];
        v8f acc = s ? a1 : a0;
#pragma unroll
        for (int v = 0; v < 8; ++v) {
            int bb = mt*16 + 8*half + v;
            float w = acc[v] + pw;
            wv[v] += w;
            float pr = w + sumr[(long)bb*V_ + n] + rvsum[bb];
            outP[((long)bb*TD + t)*V_ + n] = (dec[bb] > t) ? pr : 0.f;
        }
    }
    for (int off = 8; off >= 1; off >>= 1)
#pragma unroll
        for (int v = 0; v < 8; ++v) wv[v] += __shfl_xor(wv[v], off, 32);
    if (idx == 0)
#pragma unroll
        for (int v = 0; v < 8; ++v)
            atomicAdd(&wtsum[mt*16 + 8*half + v], wv[v]);
}

// ---------------------------------------------------------------------------
// Attention: logits -> softmax over R -> attention out + new rf (masked).
// ---------------------------------------------------------------------------
__global__ void k_attn(const float* wtsum, const float* sumv, const float* rvec,
                       const float* objs_s, const int* dec, int t,
                       float* rf, _Float16* xh, float* outA) {
    int b = blockIdx.x;
    int tid = threadIdx.x;
    __shared__ float ra[R_];
    __shared__ float inv_s;
    if (tid < R_) ra[tid] = wtsum[b] + sumv[b*R_ + tid] + rvec[b*R_ + tid];
    __syncthreads();
    if (tid == 0) {
        float mx = ra[0];
        for (int r = 1; r < R_; ++r) mx = fmaxf(mx, ra[r]);
        float s = 0.f;
        for (int r = 0; r < R_; ++r) { ra[r] = expf(ra[r] - mx); s += ra[r]; }
        inv_s = 1.f / s;
    }
    __syncthreads();
    if (tid < R_) ra[tid] *= inv_s;
    __syncthreads();
    bool m = dec[b] > t;
    if (tid < R_) outA[((long)b*TD + t)*R_ + tid] = m ? ra[tid] : 0.f;
    for (int d = tid; d < D_; d += blockDim.x) {
        float s = 0.f;
        for (int r = 0; r < R_; ++r) s += ra[r] * objs_s[((long)b*R_ + r)*D_ + d];
        float old = rf[(long)b*D_ + d];
        float nv = m ? s : old;
        rf[(long)b*D_ + d] = nv;
        xh[(long)b*KX + E_ + d] = (_Float16)nv;
    }
}

// ---------------------------------------------------------------------------
extern "C" void kernel_launch(void* const* d_in, const int* in_sizes, int n_in,
                              void* d_out, int out_size, void* d_ws, size_t ws_size,
                              hipStream_t stream) {
    const float* h0   = (const float*)d_in[0];
    const float* objs = (const float*)d_in[1];
    const int*   caps = (const int*)d_in[2];
    const int*   lens = (const int*)d_in[3];
    const float* embW = (const float*)d_in[4];
    const float* whW  = (const float*)d_in[5];
    const float* whb  = (const float*)d_in[6];
    const float* wrW  = (const float*)d_in[7];
    const float* wrb  = (const float*)d_in[8];
    const float* rhW  = (const float*)d_in[9];
    const float* rhb  = (const float*)d_in[10];
    const float* wW   = (const float*)d_in[11];
    const float* wb   = (const float*)d_in[12];
    const float* rW   = (const float*)d_in[13];
    const float* rb   = (const float*)d_in[14];
    const float* Wih  = (const float*)d_in[15];
    const float* Whh  = (const float*)d_in[16];
    const float* bih  = (const float*)d_in[17];
    const float* bhh  = (const float*)d_in[18];

    float* out  = (float*)d_out;
    float* outP = out;                               // B*TD*V
    float* outA = outP + (size_t)B_*TD*V_;           // B*TD*R
    float* outC = outA + (size_t)B_*TD*R_;           // B*TD
    float* outD = outC + (size_t)B_*TD;              // B
    float* outS = outD + B_;                         // B

    // ---- workspace carve (256B aligned bump allocator) ----
    char* p = (char*)d_ws;
    auto alloc = [&](size_t bytes) -> char* {
        char* r = p;
        p += (bytes + 255) & ~(size_t)255;
        return r;
    };
    int*       sort_ind = (int*)alloc(B_*4);
    int*       dec      = (int*)alloc(B_*4);
    int*       caps_s   = (int*)alloc((size_t)B_*T_*4);
    _Float16*  embh     = (_Float16*)alloc((size_t)V_*E_*2);
    _Float16*  whh16    = (_Float16*)alloc((size_t)H_*E_*2);
    _Float16*  wrh      = (_Float16*)alloc((size_t)D_*E_*2);
    _Float16*  rhwh     = (_Float16*)alloc((size_t)H_*D_*2);
    _Float16*  Wcat     = (_Float16*)alloc((size_t)4*H_*KX*2);
    float*     bcat     = (float*)alloc((size_t)4*H_*4);
    _Float16*  Pwh      = (_Float16*)alloc((size_t)V_*H_*2);
    _Float16*  Pwr      = (_Float16*)alloc((size_t)V_*D_*2);
    float*     Pw       = (float*)alloc((size_t)V_*4);
    float*     colsum   = (float*)alloc((size_t)D_*4);
    float*     objs_s   = (float*)alloc((size_t)B_*R_*D_*4);
    _Float16*  objs_h   = (_Float16*)alloc((size_t)B_*R_*D_*2);
    _Float16*  obj_sum  = (_Float16*)alloc((size_t)B_*D_*2);
    float*     rh_all   = (float*)alloc((size_t)B_*R_*H_*4);
    float*     r_lin    = (float*)alloc((size_t)B_*R_*4);
    float*     sumv     = (float*)alloc((size_t)B_*R_*4);
    float*     sumr     = (float*)alloc((size_t)B_*V_*4);
    _Float16*  xh       = (_Float16*)alloc((size_t)B_*KX*2);
    _Float16*  hh16     = (_Float16*)alloc((size_t)B_*H_*2);
    float*     hbuf     = (float*)alloc((size_t)B_*H_*4);
    float*     cbuf     = (float*)alloc((size_t)B_*H_*4);
    float*     rf       = (float*)alloc((size_t)B_*D_*4);
    float*     rvec     = (float*)alloc((size_t)B_*R_*4);
    float*     rvsum    = (float*)alloc((size_t)B_*4);
    float*     wtsum    = (float*)alloc((size_t)B_*4);

    // ---- setup ----
    k_sort<<<1, 64, 0, stream>>>(lens, caps, sort_ind, dec, caps_s, outC, outD, outS);
    {
        int blocks = (4*H_*KX + 255) / 256;   // covers max region
        k_convert<<<blocks, 256, 0, stream>>>(embW, whW, wrW, rhW, Wih, Whh, bih, bhh,
                                              embh, whh16, wrh, rhwh, Wcat, bcat, colsum);
    }
    k_gather<<<(B_*R_*D_)/256, 256, 0, stream>>>(h0, objs, sort_ind, objs_s, objs_h,
                                                 hbuf, cbuf, rf, xh, hh16);
    k_objsum<<<(B_*D_)/256, 256, 0, stream>>>(objs_s, obj_sum);
    k_pw<<<(V_ + 255)/256, 256, 0, stream>>>(embW, wW, wb, Pw);

    // ---- precompute GEMMs (WMMA, N-strip A reuse) ----
    // P_wh = emb @ wh^T : 750 mt x 8 strips(4), K=256
    k_gemm<4, true><<<(750*8)/4, 128, 0, stream>>>(embh, E_, whh16, E_, E_,
                                                   Pwh, H_, whb, nullptr, 8, 750*8);
    // P_wr = emb @ wr^T : 750 mt x 16 strips(4), K=256 (+ column sums)
    k_gemm<4, true><<<(750*16)/4, 128, 0, stream>>>(embh, E_, wrh, E_, E_,
                                                    Pwr, D_, wrb, colsum, 16, 750*16);
    // rh_all = objs @ rh^T : 144 mt x 8 strips(4), K=1024
    k_gemm<4, false><<<(144*8)/4, 128, 0, stream>>>(objs_h, D_, rhwh, D_, D_,
                                                    rh_all, H_, rhb, nullptr, 8, 144*8);
    // sumr_wr = obj_sum @ P_wr^T : 4 mt x 375 strips(2), K=1024
    k_gemm<2, false><<<(4*375)/4, 128, 0, stream>>>(obj_sum, D_, Pwr, D_, D_,
                                                    sumr, V_, nullptr, nullptr, 375, 4*375);
    k_rlin_sumv<<<(B_*R_ + 255)/256, 256, 0, stream>>>(objs_s, rW, rb, colsum, r_lin, sumv);

    // ---- decode loop ----
    for (int t = 0; t < TD; ++t) {
        k_prep<<<(B_*H_)/256, 256, 0, stream>>>(caps_s, embh, hh16, xh, wtsum, rvsum, t);
        k_lstm<<<32, 128, 0, stream>>>(xh, Wcat, bcat, hbuf, cbuf, hh16, dec, t);
        k_rvec<<<(B_*R_ + 255)/256, 256, 0, stream>>>(rh_all, hbuf, r_lin, rvec, rvsum);
        k_wt<<<dim3(94, 4), 128, 0, stream>>>(hh16, Pwh, Pw, sumr, rvsum, wtsum, dec, t, outP);
        k_attn<<<B_, 64, 0, stream>>>(wtsum, sumv, rvec, objs_s, dec, t, rf, xh, outA);
    }
}
